// SSMoE_Block_42425686949974
// MI455X (gfx1250) — compile-verified, matrix operand order checked by reading, not run on hardware
//
#include <hip/hip_runtime.h>
#include <hip/hip_bf16.h>
#include <math.h>
#include <stdint.h>

// ---------------------------------------------------------------------------
// Problem constants (from the reference)
// ---------------------------------------------------------------------------
static constexpr int T_TOK = 8192;   // B*S
static constexpr int DIM   = 1024;   // D
static constexpr int HID   = 4096;   // H = 4*D
static constexpr int NE    = 6;      // 4 specific + 2 shared experts

// Tiling
static constexpr int TM  = 128;
static constexpr int TN  = 128;
static constexpr int TK  = 32;
static constexpr int PAD = 40;       // LDS row stride (elems): 80 B, 16B aligned

typedef __attribute__((ext_vector_type(16))) __bf16 v16bf;
typedef __attribute__((ext_vector_type(8)))  __bf16 bf16x8;
typedef __attribute__((ext_vector_type(4)))  __bf16 bf16x4;
typedef __attribute__((ext_vector_type(8)))  float  v8f;

union Frag { v16bf v; bf16x8 h[2]; };

__device__ __forceinline__ float gelu_exact(float v) {
  return 0.5f * v * (1.0f + erff(v * 0.70710678118654752f));
}

// ---------------------------------------------------------------------------
// CDNA5 async global->LDS copy (ASYNCcnt-tracked, VFLAT op 98).
// lds_off is the wave-relative LDS byte address: per ISA 10.2 the LDS aperture
// maps flat addr[31:0] directly to the LDS logical address, so the low 32 bits
// of a flat __shared__ pointer are exactly what VDST must hold.
// ---------------------------------------------------------------------------
__device__ __forceinline__ void async_copy_b128(unsigned lds_off, const void* g) {
  asm volatile("global_load_async_to_lds_b128 %0, %1, off"
               :: "v"(lds_off), "v"(g) : "memory");
}
__device__ __forceinline__ void wait_async0() {
  asm volatile("s_wait_asynccnt 0x0" ::: "memory");
}

// ---------------------------------------------------------------------------
// Router: one wave per token. 6 logits, softmax, top-2 gates, shared probs,
// aux-loss stats via atomics. gates[t*6+e]; stats[0..3]=counts, [4..7]=probsum
// ---------------------------------------------------------------------------
__global__ __launch_bounds__(256) void moe_router(
    const float* __restrict__ x,
    const float* __restrict__ spec_rw, const float* __restrict__ spec_rb,
    const float* __restrict__ shr_rw,  const float* __restrict__ shr_rb,
    float* __restrict__ gates, float* __restrict__ stats)
{
  const int lane = threadIdx.x & 31;
  const int wv   = threadIdx.x >> 5;
  const int t    = blockIdx.x * 8 + wv;

  const float* xr = x + (size_t)t * DIM;
  float acc[6] = {0.f, 0.f, 0.f, 0.f, 0.f, 0.f};
  for (int d = lane; d < DIM; d += 32) {
    const float xv = xr[d];
    const float* rw = spec_rw + d * 4;
    acc[0] += xv * rw[0];
    acc[1] += xv * rw[1];
    acc[2] += xv * rw[2];
    acc[3] += xv * rw[3];
    const float* sw = shr_rw + d * 2;
    acc[4] += xv * sw[0];
    acc[5] += xv * sw[1];
  }
  #pragma unroll
  for (int off = 16; off > 0; off >>= 1) {
    #pragma unroll
    for (int i = 0; i < 6; ++i) acc[i] += __shfl_xor(acc[i], off, 32);
  }

  if (lane == 0) {
    float l[4];
    #pragma unroll
    for (int e2 = 0; e2 < 4; ++e2) l[e2] = acc[e2] + spec_rb[e2];
    float m = l[0];
    #pragma unroll
    for (int e2 = 1; e2 < 4; ++e2) m = fmaxf(m, l[e2]);
    float p[4], s = 0.f;
    #pragma unroll
    for (int e2 = 0; e2 < 4; ++e2) { p[e2] = __expf(l[e2] - m); s += p[e2]; }
    const float inv = 1.f / s;
    #pragma unroll
    for (int e2 = 0; e2 < 4; ++e2) p[e2] *= inv;

    int i1 = 0;
    #pragma unroll
    for (int q = 1; q < 4; ++q) if (p[q] > p[i1]) i1 = q;
    int i2 = (i1 == 0) ? 1 : 0;
    #pragma unroll
    for (int q = 0; q < 4; ++q) if (q != i1 && p[q] > p[i2]) i2 = q;

    float* gt = gates + (size_t)t * NE;
    #pragma unroll
    for (int e2 = 0; e2 < 4; ++e2)
      gt[e2] = (e2 == i1 || e2 == i2) ? p[e2] : 0.f;

    atomicAdd(&stats[i1], 1.f);
    atomicAdd(&stats[i2], 1.f);
    #pragma unroll
    for (int e2 = 0; e2 < 4; ++e2) atomicAdd(&stats[4 + e2], p[e2]);

    const float s0 = acc[4] + shr_rb[0];
    const float s1 = acc[5] + shr_rb[1];
    const float sm = fmaxf(s0, s1);
    const float e0 = __expf(s0 - sm), e1 = __expf(s1 - sm);
    const float si = 1.f / (e0 + e1);
    gt[4] = e0 * si;
    gt[5] = e1 * si;
  }
}

__global__ void moe_aux_finalize(const float* __restrict__ stats,
                                 float* __restrict__ out_aux)
{
  if (threadIdx.x == 0 && blockIdx.x == 0) {
    float aux = 0.f;
    #pragma unroll
    for (int e = 0; e < 4; ++e) {
      const float fraction = stats[e] / ((float)T_TOK * 2.0f);
      const float meanp    = stats[4 + e] / (float)T_TOK;
      aux += fraction * meanp;
    }
    out_aux[0] = 4.0f * aux;
  }
}

// out[t,d] = sum_e gate[t,e] * b2[e,d]  (initializes d_out)
__global__ __launch_bounds__(256) void moe_bias_init(
    const float* __restrict__ gates,
    const float* __restrict__ spec_b2, const float* __restrict__ shr_b2,
    float* __restrict__ out)
{
  const int i = blockIdx.x * 256 + threadIdx.x;
  const int t = i >> 10;
  const int d = i & (DIM - 1);
  const float* g = gates + (size_t)t * NE;
  float v = 0.f;
  #pragma unroll
  for (int e = 0; e < 4; ++e) v += g[e] * spec_b2[(size_t)e * DIM + d];
  #pragma unroll
  for (int e = 0; e < 2; ++e) v += g[4 + e] * shr_b2[(size_t)e * DIM + d];
  out[i] = v;
}

// ---------------------------------------------------------------------------
// f32 -> bf16 elementwise convert (vectorized)
// ---------------------------------------------------------------------------
__global__ __launch_bounds__(256) void cvt_bf16(const float* __restrict__ s,
                                                __bf16* __restrict__ d)
{
  const size_t i = ((size_t)blockIdx.x * 256 + threadIdx.x) * 4;
  const float4 f = *(const float4*)(s + i);
  bf16x4 o;
  o[0] = (__bf16)f.x; o[1] = (__bf16)f.y;
  o[2] = (__bf16)f.z; o[3] = (__bf16)f.w;
  *(bf16x4*)(d + i) = o;
}

// ---------------------------------------------------------------------------
// Tiled transpose + f32->bf16: src [z][R][C] f32 -> dst [z][C][R] bf16.
// Coalesced reads and writes via a 32x33 LDS tile. Block (32,8).
// ---------------------------------------------------------------------------
__global__ __launch_bounds__(256) void transpose_cvt(
    const float* __restrict__ src, __bf16* __restrict__ dst, int R, int C)
{
  __shared__ float tile[32][33];
  const size_t eoff = (size_t)blockIdx.z * (size_t)R * C;
  src += eoff;
  dst += eoff;
  const int bx = blockIdx.x * 32;  // col base in src
  const int by = blockIdx.y * 32;  // row base in src
  #pragma unroll
  for (int j = 0; j < 32; j += 8)
    tile[threadIdx.y + j][threadIdx.x] =
        src[(size_t)(by + threadIdx.y + j) * C + bx + threadIdx.x];
  __syncthreads();
  #pragma unroll
  for (int j = 0; j < 32; j += 8)
    dst[(size_t)(bx + threadIdx.y + j) * R + by + threadIdx.x] =
        (__bf16)tile[threadIdx.x][threadIdx.y + j];
}

// ---------------------------------------------------------------------------
// Pass 1: h_e = gate[t,e] * gelu(xb @ w1T_e^T + b1_e) -> bf16 hbuf.
// All operands bf16; tiles staged with global_load_async_to_lds_b128.
// Grid (T/128, H/128, 6); 8 waves; wave = 32x64 subtile = 2x4 WMMA frags.
// ---------------------------------------------------------------------------
__global__ __launch_bounds__(256) void moe_pass1(
    const __bf16* __restrict__ xb,
    const __bf16* __restrict__ w1T,      // [e][H][D]
    const float*  __restrict__ spec_b1, const float* __restrict__ shr_b1,
    const float*  __restrict__ gates,
    __bf16* __restrict__ hbuf)
{
  __shared__ __align__(16) __bf16 As[TM][PAD];   // [m][k]
  __shared__ __align__(16) __bf16 Bs[TN][PAD];   // [n][k]

  const int e      = blockIdx.z;
  const int tile_t = blockIdx.x * TM;
  const int tile_n = blockIdx.y * TN;

  const float* __restrict__ b1 = (e < 4)
      ? spec_b1 + (size_t)e * HID
      : shr_b1  + (size_t)(e - 4) * HID;

  const int tid  = threadIdx.x;
  const int lane = tid & 31;
  const int wv   = tid >> 5;
  const int wm   = wv & 3;
  const int wn   = wv >> 2;
  const int kh   = lane >> 4;
  const int l16  = lane & 15;

  v8f acc[2][4];
  #pragma unroll
  for (int i = 0; i < 2; ++i)
    #pragma unroll
    for (int j = 0; j < 4; ++j)
      #pragma unroll
      for (int r = 0; r < 8; ++r) acc[i][j][r] = 0.0f;

  // async staging map: wave wv copies rows [wv*16, wv*16+16) of both tiles
  const unsigned asBase = (unsigned)(uintptr_t)(&As[0][0]);
  const unsigned bsBase = (unsigned)(uintptr_t)(&Bs[0][0]);
  const int      srow   = wv * 16 + (lane >> 2);   // + i*8
  const unsigned sbyte  = (unsigned)(lane & 3) * 16u;

  const char* aG = (const char*)(xb + (size_t)tile_t * DIM);
  const char* bG = (const char*)(w1T + ((size_t)e * HID + tile_n) * DIM);

  for (int k0 = 0; k0 < DIM; k0 += TK) {
    #pragma unroll
    for (int i = 0; i < 2; ++i) {
      const int r = srow + i * 8;
      async_copy_b128(asBase + (unsigned)r * (PAD * 2) + sbyte,
                      aG + ((size_t)r * DIM + k0) * 2 + sbyte);
      async_copy_b128(bsBase + (unsigned)r * (PAD * 2) + sbyte,
                      bG + ((size_t)r * DIM + k0) * 2 + sbyte);
    }
    wait_async0();
    __syncthreads();

    Frag af[2], bf[4];
    #pragma unroll
    for (int fm = 0; fm < 2; ++fm) {
      const int row = wm * 32 + fm * 16 + l16;
      af[fm].h[0] = *(const bf16x8*)&As[row][8 * kh];        // K 0-7 / 8-15
      af[fm].h[1] = *(const bf16x8*)&As[row][16 + 8 * kh];   // K 16-23 / 24-31
    }
    #pragma unroll
    for (int fn = 0; fn < 4; ++fn) {
      const int col = wn * 64 + fn * 16 + l16;
      bf[fn].h[0] = *(const bf16x8*)&Bs[col][8 * kh];
      bf[fn].h[1] = *(const bf16x8*)&Bs[col][16 + 8 * kh];
    }
    #pragma unroll
    for (int fm = 0; fm < 2; ++fm)
      #pragma unroll
      for (int fn = 0; fn < 4; ++fn)
        acc[fm][fn] = __builtin_amdgcn_wmma_f32_16x16x32_bf16(
            false, af[fm].v, false, bf[fn].v, (short)0, acc[fm][fn],
            false, false);
    __syncthreads();
  }

  // epilogue: + b1, gelu, * gate[t,e], store bf16
  float gc[2][8];
  #pragma unroll
  for (int fm = 0; fm < 2; ++fm)
    #pragma unroll
    for (int r = 0; r < 8; ++r) {
      const int t = tile_t + wm * 32 + fm * 16 + r + 8 * kh;
      gc[fm][r] = gates[(size_t)t * NE + e];
    }
  float bc[4];
  #pragma unroll
  for (int fn = 0; fn < 4; ++fn)
    bc[fn] = b1[tile_n + wn * 64 + fn * 16 + l16];

  #pragma unroll
  for (int fm = 0; fm < 2; ++fm)
    #pragma unroll
    for (int fn = 0; fn < 4; ++fn) {
      const int col = tile_n + wn * 64 + fn * 16 + l16;
      #pragma unroll
      for (int r = 0; r < 8; ++r) {
        const int t = tile_t + wm * 32 + fm * 16 + r + 8 * kh;
        const float v = acc[fm][fn][r] + bc[fn];
        const float g = gelu_exact(v) * gc[fm][r];
        hbuf[((size_t)e * T_TOK + t) * HID + col] = (__bf16)g;
      }
    }
}

// ---------------------------------------------------------------------------
// Pass 2: out[t,d] = bias_out[t,d] + sum_e (h_e @ w2T_e^T)[t,d]
// Single accumulator set; K loop covers 6 experts x 4096. Async staging.
// ---------------------------------------------------------------------------
__global__ __launch_bounds__(256) void moe_pass2(
    const __bf16* __restrict__ hbuf,   // [e][T][H]
    const __bf16* __restrict__ w2T,    // [e][D][H]
    float* __restrict__ out)
{
  __shared__ __align__(16) __bf16 As[TM][PAD];
  __shared__ __align__(16) __bf16 Bs[TN][PAD];

  const int tile_t = blockIdx.x * TM;
  const int tile_d = blockIdx.y * TN;

  const int tid  = threadIdx.x;
  const int lane = tid & 31;
  const int wv   = tid >> 5;
  const int wm   = wv & 3;
  const int wn   = wv >> 2;
  const int kh   = lane >> 4;
  const int l16  = lane & 15;

  v8f acc[2][4];
  #pragma unroll
  for (int i = 0; i < 2; ++i)
    #pragma unroll
    for (int j = 0; j < 4; ++j)
      #pragma unroll
      for (int r = 0; r < 8; ++r) acc[i][j][r] = 0.0f;

  const unsigned asBase = (unsigned)(uintptr_t)(&As[0][0]);
  const unsigned bsBase = (unsigned)(uintptr_t)(&Bs[0][0]);
  const int      srow   = wv * 16 + (lane >> 2);
  const unsigned sbyte  = (unsigned)(lane & 3) * 16u;

  for (int e = 0; e < NE; ++e) {
    const char* aG = (const char*)(hbuf + ((size_t)e * T_TOK + tile_t) * HID);
    const char* bG = (const char*)(w2T + ((size_t)e * DIM + tile_d) * HID);

    for (int k0 = 0; k0 < HID; k0 += TK) {
      #pragma unroll
      for (int i = 0; i < 2; ++i) {
        const int r = srow + i * 8;
        async_copy_b128(asBase + (unsigned)r * (PAD * 2) + sbyte,
                        aG + ((size_t)r * HID + k0) * 2 + sbyte);
        async_copy_b128(bsBase + (unsigned)r * (PAD * 2) + sbyte,
                        bG + ((size_t)r * HID + k0) * 2 + sbyte);
      }
      wait_async0();
      __syncthreads();

      Frag af[2], bf[4];
      #pragma unroll
      for (int fm = 0; fm < 2; ++fm) {
        const int row = wm * 32 + fm * 16 + l16;
        af[fm].h[0] = *(const bf16x8*)&As[row][8 * kh];
        af[fm].h[1] = *(const bf16x8*)&As[row][16 + 8 * kh];
      }
      #pragma unroll
      for (int fn = 0; fn < 4; ++fn) {
        const int col = wn * 64 + fn * 16 + l16;
        bf[fn].h[0] = *(const bf16x8*)&Bs[col][8 * kh];
        bf[fn].h[1] = *(const bf16x8*)&Bs[col][16 + 8 * kh];
      }
      #pragma unroll
      for (int fm = 0; fm < 2; ++fm)
        #pragma unroll
        for (int fn = 0; fn < 4; ++fn)
          acc[fm][fn] = __builtin_amdgcn_wmma_f32_16x16x32_bf16(
              false, af[fm].v, false, bf[fn].v, (short)0, acc[fm][fn],
              false, false);
      __syncthreads();
    }
  }

  #pragma unroll
  for (int fm = 0; fm < 2; ++fm)
    #pragma unroll
    for (int fn = 0; fn < 4; ++fn) {
      const int col = tile_d + wn * 64 + fn * 16 + l16;
      #pragma unroll
      for (int r = 0; r < 8; ++r) {
        const int t = tile_t + wm * 32 + fm * 16 + r + 8 * kh;
        const size_t idx = (size_t)t * DIM + col;
        out[idx] = out[idx] + acc[fm][fn][r];
      }
    }
}

// ---------------------------------------------------------------------------
// Launch
// ---------------------------------------------------------------------------
extern "C" void kernel_launch(void* const* d_in, const int* in_sizes, int n_in,
                              void* d_out, int out_size, void* d_ws, size_t ws_size,
                              hipStream_t stream) {
  (void)in_sizes; (void)n_in; (void)out_size; (void)ws_size;

  const float* x       = (const float*)d_in[0];
  const float* spec_w1 = (const float*)d_in[1];
  const float* spec_b1 = (const float*)d_in[2];
  const float* spec_w2 = (const float*)d_in[3];
  const float* spec_b2 = (const float*)d_in[4];
  const float* spec_rw = (const float*)d_in[5];
  const float* spec_rb = (const float*)d_in[6];
  const float* shr_w1  = (const float*)d_in[7];
  const float* shr_b1  = (const float*)d_in[8];
  const float* shr_w2  = (const float*)d_in[9];
  const float* shr_b2  = (const float*)d_in[10];
  const float* shr_rw  = (const float*)d_in[11];
  const float* shr_rb  = (const float*)d_in[12];

  float* out = (float*)d_out;

  // workspace layout (all chunks 256B aligned)
  char*  ws    = (char*)d_ws;
  float* stats = (float*)ws;                                  // 8 f32
  float* gates = (float*)(ws + 256);                          // T*6 f32
  size_t off = 256 + (size_t)T_TOK * NE * sizeof(float);
  off = (off + 255) & ~(size_t)255;
  __bf16* xb  = (__bf16*)(ws + off); off += (size_t)T_TOK * DIM * 2;
  __bf16* w1T = (__bf16*)(ws + off); off += (size_t)NE * DIM * HID * 2;
  __bf16* w2T = (__bf16*)(ws + off); off += (size_t)NE * DIM * HID * 2;
  __bf16* hbuf = (__bf16*)(ws + off);                         // 6*T*H bf16

  hipMemsetAsync(stats, 0, 8 * sizeof(float), stream);

  moe_router<<<T_TOK / 8, 256, 0, stream>>>(
      x, spec_rw, spec_rb, shr_rw, shr_rb, gates, stats);

  moe_aux_finalize<<<1, 32, 0, stream>>>(stats, out + (size_t)T_TOK * DIM);

  moe_bias_init<<<(T_TOK * DIM) / 256, 256, 0, stream>>>(
      gates, spec_b2, shr_b2, out);

  // precision conversion + weight transposition (bf16, [e][n][k] layouts)
  cvt_bf16<<<(T_TOK * DIM / 4) / 256, 256, 0, stream>>>(x, xb);
  dim3 tb(32, 8);
  transpose_cvt<<<dim3(HID / 32, DIM / 32, 4), tb, 0, stream>>>(
      spec_w1, w1T, DIM, HID);
  transpose_cvt<<<dim3(HID / 32, DIM / 32, 2), tb, 0, stream>>>(
      shr_w1, w1T + (size_t)4 * DIM * HID, DIM, HID);
  transpose_cvt<<<dim3(DIM / 32, HID / 32, 4), tb, 0, stream>>>(
      spec_w2, w2T, HID, DIM);
  transpose_cvt<<<dim3(DIM / 32, HID / 32, 2), tb, 0, stream>>>(
      shr_w2, w2T + (size_t)4 * DIM * HID, HID, DIM);

  dim3 g1(T_TOK / TM, HID / TN, NE);
  moe_pass1<<<g1, 256, 0, stream>>>(xb, w1T, spec_b1, shr_b1, gates, hbuf);

  dim3 g2(T_TOK / TM, DIM / TN);
  moe_pass2<<<g2, 256, 0, stream>>>(hbuf, w2T, out);
}
